// DegreeGNN_77670188581370
// MI455X (gfx1250) — compile-verified, matrix operand order checked by reading, not run on hardware
//
#include <hip/hip_runtime.h>
#include <hip/hip_bf16.h>
#include <stdint.h>

// ---------------------------------------------------------------------------
// CDNA5 (gfx1250) wave32 WMMA types
// ---------------------------------------------------------------------------
typedef __attribute__((ext_vector_type(16))) __bf16 v16bf;
typedef __attribute__((ext_vector_type(8)))  float  v8f;

#define HDIM 128
#define WAVE 32
#define BN_EPS 1e-5f

// fp32 -> bf16 (round-to-nearest-even), returned in low 16 bits
__device__ __forceinline__ uint32_t f2bf(float f) {
    uint32_t u = __float_as_uint(f);
    return (u + 0x7FFFu + ((u >> 16) & 1u)) >> 16;
}
__device__ __forceinline__ uint32_t pack2bf(float lo, float hi) {
    return f2bf(lo) | (f2bf(hi) << 16);
}

// K index within a 16x32 bf16 A-tile (mirrored for B) for (lane-group g, pair p)
__device__ __forceinline__ int wmma_kbase(int g, int p) {
    return (p < 4) ? (g * 8 + 2 * p) : (16 + g * 8 + 2 * (p - 4));
}

// ---------------------------------------------------------------------------
// zero a float buffer
// ---------------------------------------------------------------------------
__global__ void k_zero_f32(float* __restrict__ p, int n) {
    int i = blockIdx.x * blockDim.x + threadIdx.x;
    if (i < n) p[i] = 0.0f;
}

// ---------------------------------------------------------------------------
// Layer-1 scalar aggregation: agg1[dst] += x[src]   (x is [N,1])
// ---------------------------------------------------------------------------
__global__ void k_agg_scalar(const int* __restrict__ src, const int* __restrict__ dst,
                             const float* __restrict__ x, float* __restrict__ agg1, int E) {
    int e = blockIdx.x * blockDim.x + threadIdx.x;
    if (e < E) atomicAdd(&agg1[dst[e]], x[src[e]]);
}

// ---------------------------------------------------------------------------
// Layer-1 post: h1 = relu(bn1(agg1[i]*Wl1[f] + x[i]*Wr1[f] + b1[f]))
// ---------------------------------------------------------------------------
__global__ void k_layer1_post(const float* __restrict__ agg1, const float* __restrict__ x,
                              const float* __restrict__ Wl1, const float* __restrict__ Wr1,
                              const float* __restrict__ b1,
                              const float* __restrict__ gn, const float* __restrict__ be,
                              const float* __restrict__ mn, const float* __restrict__ vr,
                              float* __restrict__ h1, int N) {
    int tid = blockIdx.x * blockDim.x + threadIdx.x;
    if (tid >= N * HDIM) return;
    int i = tid >> 7;
    int f = tid & (HDIM - 1);
    float val = agg1[i] * Wl1[f] + x[i] * Wr1[f] + b1[f];
    float sc  = gn[f] * rsqrtf(vr[f] + BN_EPS);
    h1[tid] = fmaxf((val - mn[f]) * sc + be[f], 0.0f);
}

// ---------------------------------------------------------------------------
// Dense aggregation: agg[dst,:] += h[src,:]   (float4 chunks, 32 threads/edge)
// ---------------------------------------------------------------------------
__global__ void k_agg_vec(const int* __restrict__ src, const int* __restrict__ dst,
                          const float* __restrict__ h, float* __restrict__ agg, int E) {
    int tid = blockIdx.x * blockDim.x + threadIdx.x;
    if (tid >= E * 32) return;
    int e = tid >> 5;
    int c = (tid & 31) << 2;   // feature base (0..124)
    int s = src[e], d = dst[e];
    const float4 v = *reinterpret_cast<const float4*>(h + (size_t)s * HDIM + c);
    float* o = agg + (size_t)d * HDIM + c;
    atomicAdd(o + 0, v.x);
    atomicAdd(o + 1, v.y);
    atomicAdd(o + 2, v.z);
    atomicAdd(o + 3, v.w);
}

// ---------------------------------------------------------------------------
// Pack fp32 activations [N,128] into bf16 WMMA A-fragments.
// frag index = ((tile*4 + kt)*32 + lane)*8 + p  == flat tid
// ---------------------------------------------------------------------------
__global__ void k_pack_rows(const float* __restrict__ h, uint32_t* __restrict__ frag,
                            int NT, int N) {
    int tid = blockIdx.x * blockDim.x + threadIdx.x;
    if (tid >= NT * 1024) return;
    int tile = tid >> 10;
    int r    = tid & 1023;
    int kt   = r >> 8;
    int lane = (r >> 3) & 31;
    int p    = r & 7;
    int g = lane >> 4, m = lane & 15;
    int row = tile * 16 + m;
    if (row >= N) row = N - 1;                      // clamp (stores are predicated later)
    int kb = kt * 32 + wmma_kbase(g, p);            // even -> 8B aligned
    const float2 v = *reinterpret_cast<const float2*>(h + (size_t)row * HDIM + kb);
    frag[tid] = pack2bf(v.x, v.y);
}

// ---------------------------------------------------------------------------
// Pack fp32 weights W[128,128] (row-major, W[k][n]) into bf16 WMMA B-fragments.
// frag index = ((ct*4 + kt)*32 + lane)*8 + p == flat tid  (8192 per matrix)
// ---------------------------------------------------------------------------
__global__ void k_pack_w(const float* __restrict__ W, uint32_t* __restrict__ frag) {
    int tid = blockIdx.x * blockDim.x + threadIdx.x;
    if (tid >= 8192) return;
    int ct   = tid >> 10;
    int r    = tid & 1023;
    int kt   = r >> 8;
    int lane = (r >> 3) & 31;
    int p    = r & 7;
    int g = lane >> 4;
    int n = ct * 16 + (lane & 15);
    int k = kt * 32 + wmma_kbase(g, p);
    frag[tid] = pack2bf(W[(size_t)k * HDIM + n], W[(size_t)(k + 1) * HDIM + n]);
}

// ---------------------------------------------------------------------------
// Fused SAGE GEMM: out = relu(bn(agg@Wl + h@Wr + bias))
// One wave per 16-row tile; loops over 8 column tiles; 8 WMMAs each.
// ---------------------------------------------------------------------------
__global__ void __launch_bounds__(256)
k_sage_gemm_bn_relu(const uint32_t* __restrict__ fragAgg, const uint32_t* __restrict__ fragH,
                    const uint32_t* __restrict__ fragWl,  const uint32_t* __restrict__ fragWr,
                    const float* __restrict__ bias,
                    const float* __restrict__ gn, const float* __restrict__ be,
                    const float* __restrict__ mn, const float* __restrict__ vr,
                    float* __restrict__ out, int NT, int N) {
    const int wave = threadIdx.x >> 5;
    const int lane = threadIdx.x & 31;
    const int tile = blockIdx.x * 8 + wave;
    if (tile >= NT) return;                         // wave-uniform: EXEC stays all-ones

    // Load this wave's A fragments (agg and h), 4 k-tiles each, 32B/lane each.
    v16bf aAgg[4], aH[4];
    #pragma unroll
    for (int kt = 0; kt < 4; ++kt) {
        size_t idx = (((size_t)tile * 4 + kt) * 32 + lane) * 8;
        aAgg[kt] = *reinterpret_cast<const v16bf*>(fragAgg + idx);
        aH[kt]   = *reinterpret_cast<const v16bf*>(fragH + idx);
    }

    const int nloc = lane & 15;
    const int g    = lane >> 4;
    const int row0 = tile * 16;

    #pragma unroll
    for (int ct = 0; ct < 8; ++ct) {
        const int n = ct * 16 + nloc;
        const float sc = gn[n] * rsqrtf(vr[n] + BN_EPS);
        const float sh = be[n] + (bias[n] - mn[n]) * sc;

        v8f c = {};
        #pragma unroll
        for (int kt = 0; kt < 4; ++kt) {
            size_t widx = (((size_t)ct * 4 + kt) * 32 + lane) * 8;
            v16bf bWl = *reinterpret_cast<const v16bf*>(fragWl + widx);
            c = __builtin_amdgcn_wmma_f32_16x16x32_bf16(
                    false, aAgg[kt], false, bWl, (short)0, c, false, false);
        }
        #pragma unroll
        for (int kt = 0; kt < 4; ++kt) {
            size_t widx = (((size_t)ct * 4 + kt) * 32 + lane) * 8;
            v16bf bWr = *reinterpret_cast<const v16bf*>(fragWr + widx);
            c = __builtin_amdgcn_wmma_f32_16x16x32_bf16(
                    false, aH[kt], false, bWr, (short)0, c, false, false);
        }

        // Epilogue: C/D layout -> lane holds column n, VGPR j holds row (g*8 + j)
        #pragma unroll
        for (int j = 0; j < 8; ++j) {
            int row = row0 + (g << 3) + j;
            if (row < N) out[(size_t)row * HDIM + n] = fmaxf(c[j] * sc + sh, 0.0f);
        }
    }
}

// ---------------------------------------------------------------------------
// Layer 4: out[i] = dot(agg[i,:], Wl4) + dot(h3[i,:], Wr4) + b4
// ---------------------------------------------------------------------------
__global__ void k_layer4(const float* __restrict__ agg, const float* __restrict__ h3,
                         const float* __restrict__ Wl4, const float* __restrict__ Wr4,
                         const float* __restrict__ b4, float* __restrict__ out, int N) {
    int i = blockIdx.x * blockDim.x + threadIdx.x;
    if (i >= N) return;
    float acc = 0.0f;
    const float* a = agg + (size_t)i * HDIM;
    const float* h = h3  + (size_t)i * HDIM;
    #pragma unroll 4
    for (int f = 0; f < HDIM; f += 4) {
        float4 av = *reinterpret_cast<const float4*>(a + f);
        float4 hv = *reinterpret_cast<const float4*>(h + f);
        acc += av.x * Wl4[f]     + hv.x * Wr4[f];
        acc += av.y * Wl4[f + 1] + hv.y * Wr4[f + 1];
        acc += av.z * Wl4[f + 2] + hv.z * Wr4[f + 2];
        acc += av.w * Wl4[f + 3] + hv.w * Wr4[f + 3];
    }
    out[i] = acc + b4[0];
}

// ---------------------------------------------------------------------------
// Launch
// ---------------------------------------------------------------------------
extern "C" void kernel_launch(void* const* d_in, const int* in_sizes, int n_in,
                              void* d_out, int out_size, void* d_ws, size_t ws_size,
                              hipStream_t stream) {
    const float* x   = (const float*)d_in[0];
    const int*   ei  = (const int*)d_in[1];
    const float* Wl1 = (const float*)d_in[2];
    const float* Wr1 = (const float*)d_in[3];
    const float* b1  = (const float*)d_in[4];
    const float* Wl2 = (const float*)d_in[5];
    const float* Wr2 = (const float*)d_in[6];
    const float* b2  = (const float*)d_in[7];
    const float* Wl3 = (const float*)d_in[8];
    const float* Wr3 = (const float*)d_in[9];
    const float* b3  = (const float*)d_in[10];
    const float* Wl4 = (const float*)d_in[11];
    const float* Wr4 = (const float*)d_in[12];
    const float* b4  = (const float*)d_in[13];
    const float* g1 = (const float*)d_in[14], *be1 = (const float*)d_in[15];
    const float* m1 = (const float*)d_in[16], *v1  = (const float*)d_in[17];
    const float* g2 = (const float*)d_in[18], *be2 = (const float*)d_in[19];
    const float* m2 = (const float*)d_in[20], *v2  = (const float*)d_in[21];
    const float* g3 = (const float*)d_in[22], *be3 = (const float*)d_in[23];
    const float* m3 = (const float*)d_in[24], *v3  = (const float*)d_in[25];

    const int N  = in_sizes[0];          // x is [N,1]
    const int E  = in_sizes[1] / 2;
    const int NT = (N + 15) / 16;        // 16-row WMMA tiles
    const int* src = ei;
    const int* dst = ei + E;
    float* out = (float*)d_out;

    // Workspace carve-out
    char* ws = (char*)d_ws;
    size_t off = 0;
    auto carve = [&](size_t bytes) -> char* {
        off = (off + 255) & ~(size_t)255;
        char* p = ws + off;
        off += bytes;
        return p;
    };
    const size_t actBytes  = (size_t)N * HDIM * sizeof(float);
    const size_t fragBytes = (size_t)NT * 1024 * sizeof(uint32_t);
    float*    hA      = (float*)carve(actBytes);
    float*    hB      = (float*)carve(actBytes);
    float*    agg     = (float*)carve(actBytes);
    uint32_t* fragAgg = (uint32_t*)carve(fragBytes);
    uint32_t* fragH   = (uint32_t*)carve(fragBytes);
    uint32_t* wf[4];
    for (int i = 0; i < 4; ++i) wf[i] = (uint32_t*)carve(8192 * sizeof(uint32_t));
    (void)ws_size;

    const int TB = 256;
    const int nAct     = N * HDIM;
    const int gAct     = (nAct + TB - 1) / TB;
    const int gEdge    = (E + TB - 1) / TB;
    const int gEdgeVec = (E * 32 + TB - 1) / TB;
    const int gPack    = (NT * 1024 + TB - 1) / TB;
    const int gGemm    = (NT + 7) / 8;
    const int gNode    = (N + TB - 1) / TB;

    // Pack weight fragments (once per launch)
    k_pack_w<<<32, TB, 0, stream>>>(Wl2, wf[0]);
    k_pack_w<<<32, TB, 0, stream>>>(Wr2, wf[1]);
    k_pack_w<<<32, TB, 0, stream>>>(Wl3, wf[2]);
    k_pack_w<<<32, TB, 0, stream>>>(Wr3, wf[3]);

    // ---- Layer 1 (IN=1 rank-1 path) ----
    k_zero_f32<<<(N + TB - 1) / TB, TB, 0, stream>>>(agg, N);
    k_agg_scalar<<<gEdge, TB, 0, stream>>>(src, dst, x, agg, E);
    k_layer1_post<<<gAct, TB, 0, stream>>>(agg, x, Wl1, Wr1, b1, g1, be1, m1, v1, hA, N);

    // ---- Layer 2 ----
    k_zero_f32<<<gAct, TB, 0, stream>>>(agg, nAct);
    k_agg_vec<<<gEdgeVec, TB, 0, stream>>>(src, dst, hA, agg, E);
    k_pack_rows<<<gPack, TB, 0, stream>>>(agg, fragAgg, NT, N);
    k_pack_rows<<<gPack, TB, 0, stream>>>(hA, fragH, NT, N);
    k_sage_gemm_bn_relu<<<gGemm, TB, 0, stream>>>(fragAgg, fragH, wf[0], wf[1],
                                                  b2, g2, be2, m2, v2, hB, NT, N);

    // ---- Layer 3 ----
    k_zero_f32<<<gAct, TB, 0, stream>>>(agg, nAct);
    k_agg_vec<<<gEdgeVec, TB, 0, stream>>>(src, dst, hB, agg, E);
    k_pack_rows<<<gPack, TB, 0, stream>>>(agg, fragAgg, NT, N);
    k_pack_rows<<<gPack, TB, 0, stream>>>(hB, fragH, NT, N);
    k_sage_gemm_bn_relu<<<gGemm, TB, 0, stream>>>(fragAgg, fragH, wf[2], wf[3],
                                                  b3, g3, be3, m3, v3, hA, NT, N);

    // ---- Layer 4 ----
    k_zero_f32<<<gAct, TB, 0, stream>>>(agg, nAct);
    k_agg_vec<<<gEdgeVec, TB, 0, stream>>>(src, dst, hA, agg, E);
    k_layer4<<<gNode, TB, 0, stream>>>(agg, hA, Wl4, Wr4, b4, out, N);
}
